// Qwen3VLVisionAttention_13932873908631
// MI455X (gfx1250) — matrix operand order repaired and checked by reading, not compiled, not run on hardware
//
#include <hip/hip_runtime.h>
#include <hip/hip_bf16.h>

typedef __attribute__((ext_vector_type(16))) _Float16 v16h;
typedef __attribute__((ext_vector_type(8)))  _Float16 v8h;
typedef __attribute__((ext_vector_type(8)))  float    v8f;

#define EMBED 1280
#define NHEAD 16
#define HDIM  80
#define HPAD  96
#define BATCH 16
#define SEQ   1024
#define MROWS (BATCH*SEQ)   /* 16384 */
#define NQKV  (3*EMBED)     /* 3840  */

union Frag { v16h v; v8h h[2]; };

__device__ __forceinline__ v8f wmma16(const Frag& a, const Frag& b, v8f c) {
    return __builtin_amdgcn_wmma_f32_16x16x32_f16(
        /*neg_a=*/false, a.v, /*neg_b=*/false, b.v,
        /*c_mod=*/(short)0, c, /*reuse_a=*/false, /*reuse_b=*/false);
}

// async global->LDS copy of 16 bytes (CDNA5 path, ASYNCcnt-tracked)
__device__ __forceinline__ void async_cp16(uint32_t lds_off, const _Float16* gaddr) {
    asm volatile("global_load_async_to_lds_b128 %0, %1, off"
                 :: "v"(lds_off), "v"(gaddr) : "memory");
}
__device__ __forceinline__ void wait_async0() {
    asm volatile("s_wait_asynccnt 0x0" ::: "memory");
}

// ---------------- helpers: convert / transpose / fill ----------------

__global__ void k_cvt(const float* __restrict__ in, _Float16* __restrict__ out, long n) {
    long i = blockIdx.x * (long)blockDim.x + threadIdx.x;
    long stride = (long)gridDim.x * blockDim.x;
    for (; i < n; i += stride) out[i] = (_Float16)in[i];
}

// in: [K][N] f32 row-major  ->  out: [N][K] f16 row-major
__global__ void k_transpose(const float* __restrict__ in, _Float16* __restrict__ out,
                            int K, int N) {
    long total = (long)K * N;
    long i = blockIdx.x * (long)blockDim.x + threadIdx.x;
    long stride = (long)gridDim.x * blockDim.x;
    for (; i < total; i += stride) {
        long n = i / K, k = i - n * K;
        out[i] = (_Float16)in[k * N + n];
    }
}

__global__ void k_fill0(uint32_t* __restrict__ p, long n) {
    long i = blockIdx.x * (long)blockDim.x + threadIdx.x;
    long stride = (long)gridDim.x * blockDim.x;
    for (; i < n; i += stride) p[i] = 0u;
}

// ---------------- 128x128 block GEMM, async-LDS staged tiles ----------------
// A: [M][K] f16 row-major, Bt: [N][K] f16 row-major (B transposed).
// Each K-step stages A-tile[128][32] and B-tile[128][32] in LDS via
// GLOBAL_LOAD_ASYNC_TO_LDS_B128 (one 64B row per thread), then 4 waves each
// compute a 64x64 sub-tile (16 WMMAs) from LDS fragments.
__device__ __forceinline__ void gemm_block128(const _Float16* __restrict__ A,
                                              const _Float16* __restrict__ Bt,
                                              int ldA, int ldB, int Mblk, int Nblk, int K,
                                              _Float16* Atile, _Float16* Btile,
                                              v8f acc[4][4]) {
    const int tid  = threadIdx.x;
    const int lane = tid & 31, wave = tid >> 5;
    const int waveM = (wave >> 1) * 64;
    const int waveN = (wave & 1) * 64;
    const int am  = lane & 15;
    const int ak  = (lane >> 4) << 3;   // A chunk offset {0,8}
    const int bk  = (lane >> 4) << 3;   // B chunk offset {0,8}

    const uint32_t aLds = (uint32_t)(uintptr_t)(Atile + tid * 32);
    const uint32_t bLds = (uint32_t)(uintptr_t)(Btile + tid * 32);
    const _Float16* aRow = A + (long)(Mblk + tid) * ldA;
    const _Float16* bRow = Bt + (long)(Nblk + tid) * ldB;

    for (int k0 = 0; k0 < K; k0 += 32) {
        __syncthreads();                      // previous tile fully consumed
#pragma unroll
        for (int c = 0; c < 4; ++c) {
            async_cp16(aLds + c * 16, aRow + k0 + c * 8);
            async_cp16(bLds + c * 16, bRow + k0 + c * 8);
        }
        wait_async0();                        // own async copies landed in LDS
        __syncthreads();                      // all waves' copies visible

        Frag af[4], bf[4];
#pragma unroll
        for (int i = 0; i < 4; ++i) {
            const v8h* p = (const v8h*)(Atile + (waveM + i * 16 + am) * 32 + ak);
            af[i].h[0] = p[0];                // K = ak   .. +7
            af[i].h[1] = p[2];                // K = 16+ak.. +7
        }
#pragma unroll
        for (int j = 0; j < 4; ++j) {
            const v8h* p = (const v8h*)(Btile + (waveN + j * 16 + am) * 32 + bk);
            bf[j].h[0] = p[0];                // K = bk .. +7
            bf[j].h[1] = p[2];                // K = 16+bk.. +7
        }
#pragma unroll
        for (int i = 0; i < 4; ++i)
#pragma unroll
            for (int j = 0; j < 4; ++j)
                acc[i][j] = wmma16(af[i], bf[j], acc[i][j]);
    }
}

// ---------------- kernel 1: fused QKV projection ----------------
__global__ __launch_bounds__(128) void k_qkv(const _Float16* __restrict__ X,
                                             const _Float16* __restrict__ WqkvT,
                                             const float* __restrict__ bias,
                                             _Float16* __restrict__ Qb,
                                             _Float16* __restrict__ Kb,
                                             _Float16* __restrict__ Vt) {
    __shared__ __align__(16) _Float16 Atile[128 * 32];
    __shared__ __align__(16) _Float16 Btile[128 * 32];
    const int lane = threadIdx.x & 31, wave = threadIdx.x >> 5;
    const int Mblk = blockIdx.y * 128, Nblk = blockIdx.x * 128;
    const int M0 = Mblk + (wave >> 1) * 64;
    const int N0 = Nblk + (wave & 1) * 64;
    v8f acc[4][4] = {};
    gemm_block128(X, WqkvT, EMBED, EMBED, Mblk, Nblk, EMBED, Atile, Btile, acc);

#pragma unroll
    for (int j = 0; j < 4; ++j) {
        const int n = N0 + j * 16 + (lane & 15);
        const float bv = bias[n];
        const int which = n / EMBED;
        const int rem = n - which * EMBED;
        const int h = rem / HDIM;
        const int d = rem - h * HDIM;
#pragma unroll
        for (int i = 0; i < 4; ++i) {
#pragma unroll
            for (int r = 0; r < 8; ++r) {
                const int m = M0 + i * 16 + r + ((lane >> 4) << 3);
                const int b = m >> 10, s = m & 1023;
                const long bh = (long)b * NHEAD + h;
                const _Float16 val = (_Float16)(acc[i][j][r] + bv);
                if (which == 0)      Qb[(bh * SEQ + s) * HPAD + d] = val;
                else if (which == 1) Kb[(bh * SEQ + s) * HPAD + d] = val;
                else                 Vt[(bh * HPAD + d) * SEQ + s] = val;
            }
        }
    }
}

// ---------------- kernel 2: attention (flash-style, no max subtraction) ----------------
__global__ __launch_bounds__(128) void k_attn(const _Float16* __restrict__ Qb,
                                              const _Float16* __restrict__ Kb,
                                              const _Float16* __restrict__ Vt,
                                              _Float16* __restrict__ Ctx) {
    __shared__ __align__(16) _Float16 stag[4][16 * 32];  // per-wave 16x32 expS staging
    const int lane = threadIdx.x & 31, wave = threadIdx.x >> 5;
    const int bh = blockIdx.y;
    const int b = bh >> 4, h = bh & 15;
    const int q0 = blockIdx.x * 64 + wave * 16;
    const _Float16* Qp = Qb + (long)bh * SEQ * HPAD;
    const _Float16* Kp = Kb + (long)bh * SEQ * HPAD;
    const _Float16* Vp = Vt + (long)bh * HPAD * SEQ;
    const float scale = 0.11180339887498948f;  // 80^-0.5
    const int g8 = (lane >> 4) << 3;           // {0,8}
    const int g16 = (lane >> 4) << 3;          // {0,8}
    _Float16* st = stag[wave];

    Frag aq[3];
    {
        const _Float16* qrow = Qp + (long)(q0 + (lane & 15)) * HPAD;
#pragma unroll
        for (int t = 0; t < 3; ++t) {
            const v8h* p = (const v8h*)(qrow + t * 32 + g8);
            aq[t].h[0] = p[0];
            aq[t].h[1] = p[2];
        }
    }

    v8f oacc[5] = {};
    float psum[8] = {};

    for (int k0 = 0; k0 < SEQ; k0 += 32) {
#pragma unroll
        for (int kt = 0; kt < 2; ++kt) {
            v8f s = {};
            const _Float16* krow = Kp + (long)(k0 + kt * 16 + (lane & 15)) * HPAD;
#pragma unroll
            for (int t = 0; t < 3; ++t) {
                Frag bk;
                const v8h* p = (const v8h*)(krow + t * 32 + g16);
                bk.h[0] = p[0];
                bk.h[1] = p[2];
                s = wmma16(aq[t], bk, s);
            }
#pragma unroll
            for (int r = 0; r < 8; ++r) {
                const float e = __expf(s[r] * scale);
                psum[r] += e;
                st[(r + g8) * 32 + kt * 16 + (lane & 15)] = (_Float16)e;
            }
        }
        Frag ap;  // relayout staged expS (same-wave DS ops are in-order)
        {
            const v8h* p = (const v8h*)(st + (lane & 15) * 32 + g8);
            ap.h[0] = p[0];
            ap.h[1] = p[2];
        }
#pragma unroll
        for (int n = 0; n < 5; ++n) {
            Frag bv;
            const v8h* p = (const v8h*)(Vp + (long)(n * 16 + (lane & 15)) * SEQ + k0 + g16);
            bv.h[0] = p[0];
            bv.h[1] = p[2];
            oacc[n] = wmma16(ap, bv, oacc[n]);
        }
    }

    float rinv[8];
#pragma unroll
    for (int r = 0; r < 8; ++r) {
        float sum = psum[r];
        sum += __shfl_xor(sum, 1);
        sum += __shfl_xor(sum, 2);
        sum += __shfl_xor(sum, 4);
        sum += __shfl_xor(sum, 8);
        rinv[r] = 1.0f / sum;
    }
#pragma unroll
    for (int n = 0; n < 5; ++n) {
        const int d = n * 16 + (lane & 15);
#pragma unroll
        for (int r = 0; r < 8; ++r) {
            const int qrow = q0 + r + g8;
            const long m = (long)b * SEQ + qrow;
            Ctx[m * EMBED + h * HDIM + d] = (_Float16)(oacc[n][r] * rinv[r]);
        }
    }
}

// ---------------- kernel 3: output projection ----------------
__global__ __launch_bounds__(128) void k_proj(const _Float16* __restrict__ Ctx,
                                              const _Float16* __restrict__ WprojT,
                                              const float* __restrict__ bias,
                                              float* __restrict__ out) {
    __shared__ __align__(16) _Float16 Atile[128 * 32];
    __shared__ __align__(16) _Float16 Btile[128 * 32];
    const int lane = threadIdx.x & 31, wave = threadIdx.x >> 5;
    const int Mblk = blockIdx.y * 128, Nblk = blockIdx.x * 128;
    const int M0 = Mblk + (wave >> 1) * 64;
    const int N0 = Nblk + (wave & 1) * 64;
    v8f acc[4][4] = {};
    gemm_block128(Ctx, WprojT, EMBED, EMBED, Mblk, Nblk, EMBED, Atile, Btile, acc);

#pragma unroll
    for (int j = 0; j < 4; ++j) {
        const int n = N0 + j * 16 + (lane & 15);
        const float bv = bias[n];
#pragma unroll
        for (int i = 0; i < 4; ++i) {
#pragma unroll
            for (int r = 0; r < 8; ++r) {
                const int m = M0 + i * 16 + r + ((lane >> 4) << 3);
                out[(long)m * EMBED + n] = acc[i][j][r] + bv;
            }
        }
    }
}

// ---------------- launch ----------------
extern "C" void kernel_launch(void* const* d_in, const int* in_sizes, int n_in,
                              void* d_out, int out_size, void* d_ws, size_t ws_size,
                              hipStream_t stream) {
    const float* hidden = (const float*)d_in[0];
    const float* w_qkv  = (const float*)d_in[1];
    const float* b_qkv  = (const float*)d_in[2];
    const float* w_proj = (const float*)d_in[3];
    const float* b_proj = (const float*)d_in[4];
    float* out = (float*)d_out;

    char* ws = (char*)d_ws;
    _Float16* Xh     = (_Float16*)ws; ws += (size_t)MROWS * EMBED * 2;              // 41.9 MB
    _Float16* WqkvT  = (_Float16*)ws; ws += (size_t)NQKV * EMBED * 2;               //  9.8 MB
    _Float16* WprojT = (_Float16*)ws; ws += (size_t)EMBED * EMBED * 2;              //  3.3 MB
    _Float16* Qb     = (_Float16*)ws; ws += (size_t)BATCH * NHEAD * SEQ * HPAD * 2; // 50.3 MB
    _Float16* Kb     = (_Float16*)ws; ws += (size_t)BATCH * NHEAD * SEQ * HPAD * 2; // 50.3 MB
    _Float16* Vt     = (_Float16*)ws; ws += (size_t)BATCH * NHEAD * HPAD * SEQ * 2; // 50.3 MB
    _Float16* Ctx    = (_Float16*)ws;                                               // 41.9 MB

    k_cvt<<<2048, 256, 0, stream>>>(hidden, Xh, (long)MROWS * EMBED);
    k_transpose<<<2048, 256, 0, stream>>>(w_qkv, WqkvT, EMBED, NQKV);
    k_transpose<<<2048, 256, 0, stream>>>(w_proj, WprojT, EMBED, EMBED);
    // zero Q and K (contiguous) so the d=80..95 pad contributes 0 to QK^T
    k_fill0<<<2048, 256, 0, stream>>>((uint32_t*)Qb,
                                      ((long)BATCH * NHEAD * SEQ * HPAD * 2 * 2) / 4);

    dim3 blk(128);
    k_qkv <<<dim3(NQKV / 128, MROWS / 128), blk, 0, stream>>>(Xh, WqkvT, b_qkv, Qb, Kb, Vt);
    k_attn<<<dim3(SEQ / 64, BATCH * NHEAD), blk, 0, stream>>>(Qb, Kb, Vt, Ctx);
    k_proj<<<dim3(EMBED / 128, MROWS / 128), blk, 0, stream>>>(Ctx, WprojT, b_proj, out);
}
